// Generator_2714419331102
// MI455X (gfx1250) — matrix operand order repaired and checked by reading, not compiled, not run on hardware
//
#include <hip/hip_runtime.h>
#include <cstdint>
#include <cstddef>

// ---------------------------------------------------------------------------
// MI455X (gfx1250) skeleton-GCN generator forward pass.
// All GEMM-shaped ops (fc, gcn 1x1 conv with adjacency folded into the A
// loader, residual 1x1 conv, temporal (7,1) conv) use v_wmma_f32_16x16x32_f16
// with f32 accumulation. Each wave computes a 16x(16*NSUB) C tile with the A
// fragment reused across NSUB B tiles; NSUB is a compile-time divisor of the
// n-tile count so the inner loop is branch-free straight-line loads + wmma.
// BN (training-mode batch stats) via LDS reduce + atomics.
// ---------------------------------------------------------------------------

typedef _Float16 half_t;
typedef __attribute__((ext_vector_type(16))) _Float16 v16h;
typedef __attribute__((ext_vector_type(8)))  _Float16 h8;
typedef __attribute__((ext_vector_type(8)))  float    v8f;
typedef __attribute__((ext_vector_type(4)))  float    f4;

#define BN_EPS 1e-5f

struct UpMapDev { int s0[32]; int s1[32]; float mult[32]; };
struct GemmGeo  { int M, N, Np, Kp, ldC, V, T, Tin, Vprev, Cin, taps, lda; };

constexpr int LD_STAGED = 0, LD_GCN = 1, LD_RES = 2, LD_TPAD = 3;
constexpr int BIAS_UNIFORM = 1, BIAS_PERW = 2;
constexpr int ACT_NONE = 0, ACT_LRELU = 1;

// ---------------- fragment loaders ----------------
// A-fragment (16x32 f16): lane<16 holds row (lane), K = [k0..k0+7, k0+16..k0+23];
// lane>=16 holds row (lane-16), K = [k0+8..k0+15, k0+24..k0+31].  (ISA 7.12.2)
__device__ __forceinline__ v16h staged_load16(const half_t* p) {
  h8 lo = *(const h8*)p;
  h8 hi = *(const h8*)(p + 16);
  v16h a;
#pragma unroll
  for (int i = 0; i < 8; ++i) { a[i] = lo[i]; a[8 + i] = hi[i]; }
  return a;
}

__device__ __forceinline__ v16h fused_load_f32(const float* p0, const float* p1, float mult) {
  f4 x0 = *(const f4*)(p0 +  0), x1 = *(const f4*)(p0 +  4);
  f4 x2 = *(const f4*)(p0 + 16), x3 = *(const f4*)(p0 + 20);
  if (p1) {
    x0 += *(const f4*)(p1 +  0); x1 += *(const f4*)(p1 +  4);
    x2 += *(const f4*)(p1 + 16); x3 += *(const f4*)(p1 + 20);
  }
  v16h a;
#pragma unroll
  for (int i = 0; i < 4; ++i) {
    a[i]      = (_Float16)(x0[i] * mult);
    a[4 + i]  = (_Float16)(x1[i] * mult);
    a[8 + i]  = (_Float16)(x2[i] * mult);
    a[12 + i] = (_Float16)(x3[i] * mult);
  }
  return a;
}

// ---------------- generic WMMA GEMM: one wave per 16x(16*NSUB) C tile --------
// C[M,N] = A[M,Kp] * Bt[Np,Kp]^T (+ taps for temporal conv) + bias, act.
// A is either staged f16, or fused on the fly from f32 activations:
//   LD_GCN : A'[m, sec*Cin+c] = imp[sec, vsrc(w), w] * hview(b, tsrc, vsrc(w), c)
//   LD_RES : A [m, c]         = hview(b, tsrc, w, c)
//   hview = upsample_s (one or two source nodes, scaled) + nearest interp_t.
template <int LOADER, int BIASM, int ACTM, int NSUB>
__global__ __launch_bounds__(256) void wmma_gemm(
    const half_t* __restrict__ Ast, const float* __restrict__ hprev,
    const half_t* __restrict__ Bt,  const float* __restrict__ bias,
    const float* __restrict__ imp,  float* __restrict__ C,
    GemmGeo g, UpMapDev um) {
  const int lane = (int)threadIdx.x & 31;
  const int wv   = (int)threadIdx.x >> 5;
  const int ntN  = (g.N + 15) >> 4;
  const int ntG  = ntN / NSUB;                 // host guarantees exact division
  const long tile = (long)blockIdx.x * 8 + wv;
  if (tile >= (long)(g.M >> 4) * ntG) return;  // M is always a multiple of 16
  const int mt = (int)(tile / ntG), ng = (int)(tile % ntG);
  const int m0 = mt << 4;
  const int n0 = ng * (NSUB * 16);
  const int rsel = lane >> 4;
  const int arow = m0 + (lane & 15);

  int w = 0, t = 0, bb = 0, tsrc = 0;
  if (LOADER != LD_STAGED) {
    w = arow % g.V;
    int tv = arow / g.V;
    t  = tv % g.T;
    bb = tv / g.T;
    tsrc = (t * g.Tin) / g.T;  // floor(i*T_in/T_out) nearest interp
  }

  v8f acc[NSUB];
#pragma unroll
  for (int s = 0; s < NSUB; ++s) acc[s] = (v8f){0.f,0.f,0.f,0.f,0.f,0.f,0.f,0.f};

  const int nsec = (LOADER == LD_GCN) ? 3 : 1;
  const int Ksec = (LOADER == LD_GCN) ? g.Cin : g.Kp;

  for (int dt = 0; dt < g.taps; ++dt) {
    // B fragment base: lane holds column n0+s*16+(lane&15); K contiguous in Bt.
    const half_t* btp0 = Bt + (size_t)dt * g.Np * g.Kp
                            + (size_t)(n0 + (lane & 15)) * g.Kp + rsel * 8;
    long tRowBase = 0;
    if (LOADER == LD_TPAD)
      tRowBase = ((long)arow + (long)bb * (6 * g.V) + (long)dt * g.V) * (long)g.Kp;

    for (int sec = 0; sec < nsec; ++sec) {
      const int kbase = sec * Ksec;  // 0 except for GCN sections
      // hoist per-section fused-loader scalars out of the k loop
      const float* p0 = nullptr; const float* p1 = nullptr; float fmult = 0.f;
      if (LOADER == LD_GCN || LOADER == LD_RES) {
        int vsrc = w;
        float scale = 1.f;
        if (LOADER == LD_GCN) {
          vsrc = (sec == 0) ? w : ((sec == 1) ? (w + g.V - 1) % g.V : (w + 1) % g.V);
          scale = imp[(sec * g.V + vsrc) * g.V + w];
        }
        const int s0 = um.s0[vsrc], s1 = um.s1[vsrc];
        fmult = um.mult[vsrc] * scale;
        const float* base = hprev + ((size_t)(bb * g.Tin + tsrc) * g.Vprev) * g.Cin
                                  + rsel * 8;
        p0 = base + (size_t)s0 * g.Cin;
        p1 = (s1 >= 0) ? (base + (size_t)s1 * g.Cin) : nullptr;
      }

      for (int c0 = 0; c0 < Ksec; c0 += 32) {
        v16h afrag;
        if (LOADER == LD_STAGED)
          afrag = staged_load16(Ast + (size_t)arow * g.lda + c0 + rsel * 8);
        else if (LOADER == LD_TPAD)
          afrag = staged_load16(Ast + tRowBase + c0 + rsel * 8);
        else
          afrag = fused_load_f32(p0 + c0, p1 ? (p1 + c0) : nullptr, fmult);

#pragma unroll
        for (int s = 0; s < NSUB; ++s) {
          v16h bfrag = staged_load16(btp0 + (size_t)s * 16 * g.Kp + kbase + c0);
          acc[s] = __builtin_amdgcn_wmma_f32_16x16x32_f16(false, afrag, false, bfrag,
                                                          (short)0, acc[s], false, false);
        }
      }
    }
  }

#pragma unroll
  for (int s = 0; s < NSUB; ++s) {
    const int col = n0 + s * 16 + (lane & 15);
    if (col < g.N) {
      float bu = (BIASM == BIAS_UNIFORM) ? bias[col] : 0.f;
#pragma unroll
      for (int r = 0; r < 8; ++r) {
        int row = m0 + r + rsel * 8;  // C/D layout: M = vgpr + 8*lane[4]
        if (row < g.M) {
          float v = acc[s][r] + bu;
          if (BIASM == BIAS_PERW) v += bias[(row % g.V) * g.N + col];
          if (ACTM == ACT_LRELU) v = (v >= 0.f) ? v : 0.2f * v;
          C[(size_t)row * g.ldC + col] = v;
        }
      }
    }
  }
}

// ---------------- staging / pack kernels ----------------
// Weights -> f16, stored as Bt [Np x KpSec] (row n = output channel n), zero padded.
__global__ void pack_bt_kernel(const float* __restrict__ src, half_t* __restrict__ dst,
                               int N, int K, int srcRow, int srcCol, long srcOff,
                               int KpSec, int dstRow, long dstOff, long total) {
  long i = (long)blockIdx.x * 256 + threadIdx.x;
  if (i >= total) return;
  int k = (int)(i % KpSec);
  int n = (int)(i / KpSec);
  float v = (n < N && k < K) ? src[srcOff + (long)n * srcRow + (long)k * srcCol] : 0.f;
  dst[dstOff + (long)n * dstRow + k] = (half_t)v;
}

// fc input: concat(label_emb[labels], x) -> f16 padded to Kp=608
__global__ void stage_fc_kernel(const float* __restrict__ x, const int* __restrict__ labels,
                                const float* __restrict__ emb, half_t* __restrict__ A,
                                int M, int Kp) {
  long i = (long)blockIdx.x * 256 + threadIdx.x;
  if (i >= (long)M * Kp) return;
  int k = (int)(i % Kp);
  int m = (int)(i / Kp);
  float v = 0.f;
  if (k < 60)       v = emb[labels[m] * 60 + k];
  else if (k < 572) v = x[(long)m * 512 + (k - 60)];
  A[i] = (half_t)v;
}

// zero-pad the gcn output in T (3 each side) into an f16 buffer [B,(T+6),V,Kt]
__global__ void stage_tpad_kernel(const float* __restrict__ yg, half_t* __restrict__ tpad,
                                  int B_, int T, int V, int cout, int Kt, long total) {
  long i = (long)blockIdx.x * 256 + threadIdx.x;
  if (i >= total) return;
  int k = (int)(i % Kt);
  long r = i / Kt;
  int v = (int)(r % V);
  long bt = r / V;
  int tp = (int)(bt % (T + 6));
  int b  = (int)(bt / (T + 6));
  int t = tp - 3;
  float val = 0.f;
  if (t >= 0 && t < T && k < cout)
    val = yg[(((long)b * T + t) * V + v) * cout + k];
  tpad[i] = (half_t)val;
}

// adjacency-contracted gcn bias: biasW[w,o] = sum_k imp[k,vsrc_k(w),w]*gb[k*cout+o]
__global__ void gcn_bias_kernel(const float* __restrict__ imp, const float* __restrict__ gb,
                                float* __restrict__ biasW, int V, int cout) {
  int i = blockIdx.x * 256 + threadIdx.x;
  if (i >= V * cout) return;
  int o = i % cout, w = i / cout;
  float s = 0.f;
  for (int k = 0; k < 3; ++k) {
    int vs = (k == 0) ? w : ((k == 1) ? (w + V - 1) % V : (w + 1) % V);
    s += imp[(k * V + vs) * V + w] * gb[k * cout + o];
  }
  biasW[i] = s;
}

__global__ void zero_kernel(float* __restrict__ p, int n) {
  int i = blockIdx.x * 256 + threadIdx.x;
  if (i < n) p[i] = 0.f;
}

// per-channel sum / sumsq for Y (and optionally R), LDS tree reduce + atomics
__global__ __launch_bounds__(256) void stats_kernel(const float* __restrict__ Y,
                                                    const float* __restrict__ R,
                                                    float* __restrict__ stats,
                                                    long M, int cout) {
  const int ch = blockIdx.x;
  const int nch = gridDim.y;
  float sy = 0.f, syy = 0.f, sr = 0.f, srr = 0.f;
  for (long m = (long)blockIdx.y * 256 + threadIdx.x; m < M; m += (long)nch * 256) {
    float v = Y[m * cout + ch]; sy += v; syy += v * v;
    if (R) { float u = R[m * cout + ch]; sr += u; srr += u * u; }
  }
  __shared__ float sm[512];
  sm[threadIdx.x] = sy; sm[256 + threadIdx.x] = syy;
  __syncthreads();
  for (int s = 128; s > 0; s >>= 1) {
    if (threadIdx.x < s) { sm[threadIdx.x] += sm[threadIdx.x + s];
                           sm[256 + threadIdx.x] += sm[256 + threadIdx.x + s]; }
    __syncthreads();
  }
  if (threadIdx.x == 0) { atomicAdd(&stats[ch], sm[0]); atomicAdd(&stats[cout + ch], sm[256]); }
  if (R) {
    __syncthreads();
    sm[threadIdx.x] = sr; sm[256 + threadIdx.x] = srr;
    __syncthreads();
    for (int s = 128; s > 0; s >>= 1) {
      if (threadIdx.x < s) { sm[threadIdx.x] += sm[threadIdx.x + s];
                             sm[256 + threadIdx.x] += sm[256 + threadIdx.x + s]; }
      __syncthreads();
    }
    if (threadIdx.x == 0) { atomicAdd(&stats[2 * cout + ch], sm[0]);
                            atomicAdd(&stats[3 * cout + ch], sm[256]); }
  }
}

// H = lrelu( BN(Y) + BN(R) )   (R==null -> no residual term)
__global__ void finalize_kernel(const float* __restrict__ Y, const float* __restrict__ R,
                                const float* __restrict__ stats,
                                const float* __restrict__ g_, const float* __restrict__ b_,
                                const float* __restrict__ rg, const float* __restrict__ rb,
                                float* __restrict__ H, long M, int cout) {
  long i = (long)blockIdx.x * 256 + threadIdx.x;
  if (i >= M * cout) return;
  int o = (int)(i % cout);
  float invM = 1.f / (float)M;
  float mY = stats[o] * invM;
  float vY = stats[cout + o] * invM - mY * mY;
  float y = g_[o] * (Y[i] - mY) * rsqrtf(vY + BN_EPS) + b_[o];
  if (R) {
    float mR = stats[2 * cout + o] * invM;
    float vR = stats[3 * cout + o] * invM - mR * mR;
    y += rg[o] * (R[i] - mR) * rsqrtf(vR + BN_EPS) + rb[o];
  }
  H[i] = (y >= 0.f) ? y : 0.2f * y;
}

// plain VALU temporal conv for cout==3 blocks
__global__ void tconv_valu_kernel(const float* __restrict__ yg, const float* __restrict__ tw,
                                  const float* __restrict__ tb, float* __restrict__ yt,
                                  long M, int V, int T, int cout) {
  long i = (long)blockIdx.x * 256 + threadIdx.x;
  if (i >= M * cout) return;
  int o = (int)(i % cout);
  long m = i / cout;
  int t = (int)((m / V) % T);
  float acc = tb[o];
  for (int dt = 0; dt < 7; ++dt) {
    int ts = t + dt - 3;
    if (ts < 0 || ts >= T) continue;
    long sm = m + (long)(dt - 3) * V;
    for (int c = 0; c < cout; ++c)
      acc += yg[sm * cout + c] * tw[(o * cout + c) * 7 + dt];
  }
  yt[i] = acc;
}

// block-5 helpers (cin=cout=3): explicit f32 hview, VALU gcn, tanh finalize -> NCHW out
__global__ void stage_hview_kernel(const float* __restrict__ hp, float* __restrict__ hv,
                                   UpMapDev um, int B_, int T, int V, int Tin, int Vprev,
                                   int C, long total) {
  long i = (long)blockIdx.x * 256 + threadIdx.x;
  if (i >= total) return;
  int c = (int)(i % C);
  long r = i / C;
  int w = (int)(r % V);
  long bt = r / V;
  int t = (int)(bt % T);
  int b = (int)(bt / T);
  int ts = (t * Tin) / T;
  const float* base = hp + ((size_t)(b * Tin + ts) * Vprev) * C;
  float v = base[um.s0[w] * C + c];
  if (um.s1[w] >= 0) v += base[um.s1[w] * C + c];
  hv[i] = v * um.mult[w];
}

__global__ void gcn_valu_kernel(const float* __restrict__ hv, const float* __restrict__ imp,
                                const float* __restrict__ gw, const float* __restrict__ gb,
                                float* __restrict__ yg, long M, int V, int C, int cout) {
  long i = (long)blockIdx.x * 256 + threadIdx.x;
  if (i >= M * cout) return;
  int o = (int)(i % cout);
  long m = i / cout;
  int w = (int)(m % V);
  long rowbt = m - w;
  float acc = 0.f;
  for (int k = 0; k < 3; ++k) {
    int vs = (k == 0) ? w : ((k == 1) ? (w + V - 1) % V : (w + 1) % V);
    float s = imp[(k * V + vs) * V + w];
    float d = gb[k * cout + o];
    const float* p = hv + (rowbt + vs) * C;
    for (int c = 0; c < C; ++c) d += p[c] * gw[(k * cout + o) * C + c];
    acc += s * d;
  }
  yg[i] = acc;
}

__global__ void finalize_tanh_kernel(const float* __restrict__ Y, const float* __restrict__ hv,
                                     const float* __restrict__ stats,
                                     const float* __restrict__ g_, const float* __restrict__ b_,
                                     float* __restrict__ out, long M, int cout, int T, int V) {
  long i = (long)blockIdx.x * 256 + threadIdx.x;
  if (i >= M * cout) return;
  int o = (int)(i % cout);
  long m = i / cout;
  int w = (int)(m % V);
  long bt = m / V;
  int t = (int)(bt % T);
  int b = (int)(bt / T);
  float invM = 1.f / (float)M;
  float mY = stats[o] * invM;
  float vY = stats[cout + o] * invM - mY * mY;
  float y = g_[o] * (Y[i] - mY) * rsqrtf(vY + BN_EPS) + b_[o] + hv[i];
  out[(((long)b * cout + o) * T + t) * V + w] = tanhf(y);
}

// ---------------------------------------------------------------------------
// host side
// ---------------------------------------------------------------------------
static inline long divUpL(long a, long b) { return (a + b - 1) / b; }
static inline int nsubFor(int N) {
  int ntN = (N + 15) / 16;
  if (ntN % 4 == 0) return 4;
  if (ntN % 2 == 0) return 2;
  return 1;
}
static inline long gemmBlocks(long M, int N) {
  int ntN = (N + 15) / 16;
  int ntG = ntN / nsubFor(N);
  long tiles = (M / 16) * ntG;
  return divUpL(tiles, 8);  // 8 waves per 256-thread workgroup
}

template <int LOADER, int BIASM, int ACTM>
static void launch_gemm(hipStream_t stream, const half_t* Ast, const float* hprev,
                        const half_t* Bt, const float* bias, const float* imp, float* C,
                        const GemmGeo& g, const UpMapDev& um) {
  long blocks = gemmBlocks(g.M, g.N);
  int ns = nsubFor(g.N);
  if (ns == 4)
    wmma_gemm<LOADER, BIASM, ACTM, 4><<<blocks, 256, 0, stream>>>(Ast, hprev, Bt, bias, imp, C, g, um);
  else if (ns == 2)
    wmma_gemm<LOADER, BIASM, ACTM, 2><<<blocks, 256, 0, stream>>>(Ast, hprev, Bt, bias, imp, C, g, um);
  else
    wmma_gemm<LOADER, BIASM, ACTM, 1><<<blocks, 256, 0, stream>>>(Ast, hprev, Bt, bias, imp, C, g, um);
}

extern "C" void kernel_launch(void* const* d_in, const int* in_sizes, int n_in,
                              void* d_out, int out_size, void* d_ws, size_t ws_size,
                              hipStream_t stream) {
  (void)in_sizes; (void)n_in; (void)out_size; (void)ws_size;
  const int BATCH = 2048;

  // ---- param pointers (jax tree_leaves: sorted dict keys) ----
  // order: x, labels, blocks[0..4]{bnb,bng,gb,gw,imp,rb,rbnb,rbng,rw,tb,tw},
  //        blocks[5]{bnb,bng,gb,gw,imp,tb,tw}, fc_b, fc_w, label_emb
  auto F = [&](int i) { return (const float*)d_in[i]; };
  const float* x      = F(0);
  const int*   labels = (const int*)d_in[1];
  struct BP { const float *bnb,*bng,*gb,*gw,*imp,*rb,*rbnb,*rbng,*rw,*tb,*tw; };
  BP bp[6];
  for (int i = 0; i < 5; ++i) {
    int b0 = 2 + 11 * i;
    bp[i] = { F(b0+0),F(b0+1),F(b0+2),F(b0+3),F(b0+4),F(b0+5),
              F(b0+6),F(b0+7),F(b0+8),F(b0+9),F(b0+10) };
  }
  bp[5] = { F(57),F(58),F(59),F(60),F(61),nullptr,nullptr,nullptr,nullptr,F(62),F(63) };
  const float* fc_b = F(64);
  const float* fc_w = F(65);
  const float* emb  = F(66);

  // ---- workspace bump allocator ----
  size_t off = 0;
  auto alloc = [&](size_t bytes) -> void* {
    off = (off + 255) & ~(size_t)255;
    void* p = (char*)d_ws + off;
    off += bytes;
    return p;
  };
  half_t* Afc    = (half_t*)alloc((size_t)2048 * 608 * 2);
  half_t* fcBt   = (half_t*)alloc((size_t)512 * 608 * 2);
  half_t* gcnBt  = (half_t*)alloc((size_t)256 * 1536 * 2);
  half_t* resBt  = (half_t*)alloc((size_t)256 * 512 * 2);
  half_t* tBt    = (half_t*)alloc((size_t)7 * 256 * 256 * 2);
  float*  biasW  = (float*)alloc(1024 * 4);
  float*  stats  = (float*)alloc(1024 * 4);
  const size_t MAXMC = 25165824;  // max M*cout (block 3)
  float*  YG = (float*)alloc(MAXMC * 4);
  float*  YT = (float*)alloc(MAXMC * 4);
  float*  RB = (float*)alloc(MAXMC * 4);
  half_t* TPAD = (half_t*)alloc((size_t)2048 * 38 * 12 * 32 * 2);  // block-3 max
  float*  H[2];
  H[0] = (float*)alloc(MAXMC * 4);
  H[1] = (float*)alloc(MAXMC * 4);

  // ---- upsample maps: val(w) = (h[s0] + (s1>=0? h[s1])) * mult ----
  auto identMap = [](int V) { UpMapDev u{}; for (int i = 0; i < V; ++i) { u.s0[i]=i; u.s1[i]=-1; u.mult[i]=1.f; } return u; };
  UpMapDev um6{};   // lvl2: 1 -> 6, scale 0.5
  for (int i = 0; i < 6; ++i) { um6.s0[i]=0; um6.s1[i]=-1; um6.mult[i] = (i==0)?1.f:0.5f; }
  UpMapDev um12{};  // lvl1: 6 -> 12, interleaved means
  for (int j = 0; j < 6; ++j) {
    um12.s0[2*j]=j;   um12.s1[2*j]=-1;        um12.mult[2*j]=1.f;
    um12.s0[2*j+1]=j; um12.s1[2*j+1]=(j+1)%6; um12.mult[2*j+1]=0.5f;
  }
  UpMapDev um25{};  // lvl0: 12 -> 25
  for (int j = 0; j < 12; ++j) {
    um25.s0[2*j]=j;   um25.s1[2*j]=-1;         um25.mult[2*j]=1.f;
    um25.s0[2*j+1]=j; um25.s1[2*j+1]=(j+1)%12; um25.mult[2*j+1]=0.5f;
  }
  um25.s0[24]=0; um25.s1[24]=6; um25.mult[24]=0.5f;

  auto packBt = [&](const float* src, half_t* dst, int N, int K, int srcRow, int srcCol,
                    long srcOff, int KpSec, int dstRow, long dstOff) {
    long total = (long)((N + 15) & ~15) * KpSec;
    pack_bt_kernel<<<divUpL(total, 256), 256, 0, stream>>>(
        src, dst, N, K, srcRow, srcCol, srcOff, KpSec, dstRow, dstOff, total);
  };

  // ---- fc layer: (2048 x 608) x (608 x 512), +bias, lrelu ----
  {
    long tot = 2048L * 608;
    stage_fc_kernel<<<divUpL(tot, 256), 256, 0, stream>>>(x, labels, emb, Afc, 2048, 608);
    packBt(fc_w, fcBt, 512, 572, 572, 1, 0, 608, 608, 0);
    GemmGeo g = {2048, 512, 512, 608, 512, 1, 1, 1, 1, 512, 1, 608};
    launch_gemm<LD_STAGED, BIAS_UNIFORM, ACT_LRELU>(stream, Afc, nullptr, fcBt, fc_b,
                                                    nullptr, H[0], g, identMap(1));
  }

  // ---- blocks 0..4 ----
  struct Cfg { int T, V, Tin, Vprev, Cin, cout; const UpMapDev* um; };
  UpMapDev id1 = identMap(1), id6 = identMap(6), id12 = identMap(12);
  Cfg cfgs[5] = {
    { 8,  1,  1,  1, 512, 256, &id1  },
    { 8,  6,  8,  1, 256, 128, &um6  },
    { 16, 6,  8,  6, 128, 64,  &id6  },
    { 32, 12, 16, 6, 64,  32,  &um12 },
    { 64, 12, 32, 12, 32, 3,   &id12 },
  };

  int cur = 0;
  for (int i = 0; i < 5; ++i) {
    const Cfg& c = cfgs[i];
    const BP& p = bp[i];
    const long M = (long)BATCH * c.T * c.V;
    const int Np = (c.cout + 15) & ~15;
    const float* hprev = H[cur];
    float* hnext = H[1 - cur];

    // stage weights: gcn Bt [Np x 3Cin] (3 sections), res Bt [Np x Cin], temporal Bt [7][Np][cout]
    for (int k = 0; k < 3; ++k)
      packBt(p.gw, gcnBt, c.cout, c.Cin, c.Cin, 1, (long)k * c.cout * c.Cin,
             c.Cin, 3 * c.Cin, (long)k * c.Cin);
    packBt(p.rw, resBt, c.cout, c.Cin, c.Cin, 1, 0, c.Cin, c.Cin, 0);
    if (c.cout >= 32)
      for (int dt = 0; dt < 7; ++dt)
        packBt(p.tw, tBt, c.cout, c.cout, c.cout * 7, 7, dt, c.cout, c.cout,
               (long)dt * Np * c.cout);
    gcn_bias_kernel<<<divUpL((long)c.V * c.cout, 256), 256, 0, stream>>>(
        p.imp, p.gb, biasW, c.V, c.cout);

    // gcn GEMM (adjacency + upsample + interp fused into A loader), per-node bias
    {
      GemmGeo g = {(int)M, c.cout, Np, 3 * c.Cin, c.cout,
                   c.V, c.T, c.Tin, c.Vprev, c.Cin, 1, 0};
      launch_gemm<LD_GCN, BIAS_PERW, ACT_NONE>(stream, nullptr, hprev, gcnBt, biasW,
                                               p.imp, YG, g, *c.um);
    }
    // residual GEMM on the same (upsampled+interpolated) view
    {
      GemmGeo g = {(int)M, c.cout, Np, c.Cin, c.cout,
                   c.V, c.T, c.Tin, c.Vprev, c.Cin, 1, 0};
      launch_gemm<LD_RES, BIAS_UNIFORM, ACT_NONE>(stream, nullptr, hprev, resBt, p.rb,
                                                  nullptr, RB, g, *c.um);
    }
    // temporal (7,1) conv
    if (c.cout >= 32) {
      long padTot = (long)BATCH * (c.T + 6) * c.V * c.cout;
      stage_tpad_kernel<<<divUpL(padTot, 256), 256, 0, stream>>>(
          YG, TPAD, BATCH, c.T, c.V, c.cout, c.cout, padTot);
      GemmGeo g = {(int)M, c.cout, Np, c.cout, c.cout,
                   c.V, c.T, c.T, c.V, c.cout, 7, 0};
      launch_gemm<LD_TPAD, BIAS_UNIFORM, ACT_NONE>(stream, TPAD, nullptr, tBt, p.tb,
                                                   nullptr, YT, g, *c.um);
    } else {
      tconv_valu_kernel<<<divUpL(M * c.cout, 256), 256, 0, stream>>>(
          YG, p.tw, p.tb, YT, M, c.V, c.T, c.cout);
    }
    // batch-norm stats for YT and RB, then fuse BN + BN(res) + lrelu
    zero_kernel<<<1, 256, 0, stream>>>(stats, 4 * c.cout);
    stats_kernel<<<dim3(c.cout, 64), 256, 0, stream>>>(YT, RB, stats, M, c.cout);
    finalize_kernel<<<divUpL(M * c.cout, 256), 256, 0, stream>>>(
        YT, RB, stats, p.bng, p.bnb, p.rbng, p.rbnb, hnext, M, c.cout);
    cur = 1 - cur;
  }

  // ---- block 5: lvl0, 3->3, upsample 12->25, T 64->128, res = h, tanh ----
  {
    const BP& p = bp[5];
    const int T = 128, V = 25, Tin = 64, Vprev = 12, C = 3;
    const long M = (long)BATCH * T * V;
    float* HV = H[1 - cur];
    long tot = M * C;
    stage_hview_kernel<<<divUpL(tot, 256), 256, 0, stream>>>(
        H[cur], HV, um25, BATCH, T, V, Tin, Vprev, C, tot);
    gcn_valu_kernel<<<divUpL(M * C, 256), 256, 0, stream>>>(
        HV, p.imp, p.gw, p.gb, YG, M, V, C, C);
    tconv_valu_kernel<<<divUpL(M * C, 256), 256, 0, stream>>>(
        YG, p.tw, p.tb, YT, M, V, T, C);
    zero_kernel<<<1, 256, 0, stream>>>(stats, 4 * C);
    stats_kernel<<<dim3(C, 64), 256, 0, stream>>>(YT, nullptr, stats, M, C);
    finalize_tanh_kernel<<<divUpL(M * C, 256), 256, 0, stream>>>(
        YT, HV, stats, p.bng, p.bnb, (float*)d_out, M, C, T, V);
  }
}
